// SelfOtherAwareAttention_77618648973566
// MI455X (gfx1250) — matrix operand order repaired and checked by reading, not compiled, not run on hardware
//
#include <hip/hip_runtime.h>
#include <hip/hip_bf16.h>

// Shapes: B=4, L=S=2048, D=256, H=8, hd=32
#define L_DIM 2048
#define S_DIM 2048
#define D_DIM 256
#define H_NUM 8
#define HD    32
#define S_TILES (S_DIM / 16)
#define PLANE (8192ull * 256ull)   // one projection plane, elements

// LDS staging: 16 rows x 512B data + 16B pad per row (TDM pad feature)
#define ROWB      528
#define PLANE_LDS (16 * ROWB)      // 8448 B per k-plane tile

#if defined(__has_builtin)
#if __has_builtin(__builtin_amdgcn_tensor_load_to_lds) && __has_builtin(__builtin_amdgcn_s_wait_tensorcnt)
#define HAVE_TDM 1
#endif
#endif
#ifndef HAVE_TDM
#define HAVE_TDM 0
#endif

typedef __attribute__((ext_vector_type(16))) __bf16 v16bf;
typedef __attribute__((ext_vector_type(8)))  float  v8f;
typedef __attribute__((ext_vector_type(8)))  __bf16 bf16x8;
typedef __attribute__((ext_vector_type(4)))  float  f4;
typedef __attribute__((ext_vector_type(4)))  unsigned int u32x4;
typedef __attribute__((ext_vector_type(8)))  int i32x8;
typedef __attribute__((ext_vector_type(4)))  int i32x4;

union V16U { v16bf v; __bf16 e[16]; bf16x8 h[2]; };

__device__ __forceinline__ v8f wmma_bf16(v16bf a, v16bf b, v8f c) {
  // D = A x B + C, 16x16x32 bf16 -> f32
  return __builtin_amdgcn_wmma_f32_16x16x32_bf16(false, a, false, b, (short)0, c, false, false);
}

// ---- f32 -> (hi,lo) bf16 split tile loaders (projection GEMM) ----
// A layout (16-bit A 16x32): lane m = lane&15; per-lane K = kh*16 + 0..15 (contiguous)
// B layout (16-bit B 32x16): lane n = lane&15; per-lane K = kh*16 + 0..15 (contiguous)
__device__ __forceinline__ void split_tile(const float* p0, const float* p1,
                                           v16bf& hiv, v16bf& lov) {
  f4 a0 = *(const f4*)(p0);
  f4 a1 = *(const f4*)(p0 + 4);
  f4 b0 = *(const f4*)(p1);
  f4 b1 = *(const f4*)(p1 + 4);
  float v[16];
#pragma unroll
  for (int i = 0; i < 4; ++i) {
    v[i] = a0[i]; v[4 + i] = a1[i]; v[8 + i] = b0[i]; v[12 + i] = b1[i];
  }
  V16U hu, lu;
#pragma unroll
  for (int i = 0; i < 16; ++i) {
    float x = v[i];
    __bf16 hb = (__bf16)x;
    hu.e[i] = hb;
    lu.e[i] = (__bf16)(x - (float)hb);
  }
  hiv = hu.v; lov = lu.v;
}

// ---- bf16 fragment loaders ----
__device__ __forceinline__ v16bf load_bf_a(const __bf16* rowbase, int lane) {
  const __bf16* p = rowbase + (size_t)(lane & 15) * D_DIM + ((lane >> 4) << 4);
  V16U u;
  u.h[0] = *(const bf16x8*)(p);        // K = kh*16 + 0..7
  u.h[1] = *(const bf16x8*)(p + 8);    // K = kh*16 + 8..15
  return u.v;
}
__device__ __forceinline__ v16bf load_bf_b(const __bf16* rowbase, int lane) {
  const __bf16* p = rowbase + (size_t)(lane & 15) * D_DIM + ((lane >> 4) << 4);
  V16U u;
  u.h[0] = *(const bf16x8*)(p);        // K = kh*16 + 0..7
  u.h[1] = *(const bf16x8*)(p + 8);    // K = kh*16 + 8..15
  return u.v;
}
#if HAVE_TDM
// B fragment from the TDM-staged LDS tile (row stride 528B -> conflict-free)
__device__ __forceinline__ v16bf lds_b_frag(const unsigned char* plane, int lane, int h) {
  const unsigned char* p = plane + (lane & 15) * ROWB + h * 64 + ((lane >> 4) << 5);
  V16U u;
  u.h[0] = *(const bf16x8*)(p);        // ds_load_b128
  u.h[1] = *(const bf16x8*)(p + 16);
  return u.v;
}

// Issue one TDM: 16 rows x 256 bf16 cols, global -> LDS with 16B/row padding.
// D# per ISA 8.3/8.4: count=1, type=2(image), data_size=2B,
// pad_enable=1, pad_interval=6 (128 dwords = 512B), pad_amount=3 (4 dwords = 16B)
__device__ __forceinline__ void tdm_load_tile(unsigned lds_off, const __bf16* gptr) {
  unsigned long long ga = (unsigned long long)(uintptr_t)gptr;
  u32x4 g0;
  g0[0] = 1u;                                             // count=1
  g0[1] = lds_off;                                        // lds_addr (bytes)
  g0[2] = (unsigned)ga;                                   // global_addr[31:0]
  g0[3] = (unsigned)((ga >> 32) & 0x01FFFFFFu) | (2u << 30); // addr[56:32] | type=2
  i32x8 g1;
  g1[0] = (int)((1u << 16) | (1u << 20) | (6u << 22) | (3u << 25)); // dsz=2B,pad
  g1[1] = (int)(256u << 16);   // tensor_dim0[15:0]=256 in bits 63:48
  g1[2] = (int)(8192u << 16);  // tensor_dim1[15:0]=8192 in bits 111:96... (bits 95:80)
  g1[3] = (int)(256u << 16);   // tile_dim0=256 in bits 127:112
  g1[4] = 16;                  // tile_dim1=16, tile_dim2=0
  g1[5] = 256;                 // tensor_dim0_stride=256 (elements)
  g1[6] = 0;
  g1[7] = 0;
  i32x4 z4 = {0, 0, 0, 0};
#if defined(__clang_major__) && __clang_major__ >= 23
  i32x8 z8 = {0, 0, 0, 0, 0, 0, 0, 0};
  __builtin_amdgcn_tensor_load_to_lds(g0, g1, z4, z4, z8, 0);
#else
  __builtin_amdgcn_tensor_load_to_lds(g0, g1, z4, z4, 0);
#endif
}
#endif  // HAVE_TDM

// ============================================================================
// Kernel 1: four projections out = (X @ W^T) * scale -> bf16 hi/lo planes in ws
// plane order: [qs_hi, qs_lo, qo_hi, qo_lo, ks_hi, ks_lo, ko_hi, ko_lo]
// ============================================================================
__global__ __launch_bounds__(256) void proj_kernel(
    const float* __restrict__ q, const float* __restrict__ kmat,
    const float* __restrict__ Wqs, const float* __restrict__ Wqo,
    const float* __restrict__ Wks, const float* __restrict__ Wko,
    __bf16* __restrict__ ws) {
  int lane = threadIdx.x & 31;
  int wave = (blockIdx.x << 3) + (threadIdx.x >> 5);  // 0..32767
  int proj = wave >> 13;                              // 0..3
  int t    = wave & 8191;
  int m0   = (t >> 4) << 4;
  int n0   = (t & 15) << 4;

  const float* A = (proj < 2) ? q : kmat;
  const float* W = (proj == 0) ? Wqs : (proj == 1) ? Wqo : (proj == 2) ? Wks : Wko;
  float scale = (proj < 2) ? 0.17677669529663687f : 1.0f;  // hd^-0.5 for q projs
  __bf16* hi_plane = ws + (size_t)proj * 2 * PLANE;
  __bf16* lo_plane = hi_plane + PLANE;

  const float* arow = A + (size_t)(m0 + (lane & 15)) * D_DIM + ((lane >> 4) << 4);
  const float* brow = W + (size_t)(n0 + (lane & 15)) * D_DIM + ((lane >> 4) << 4);

  v8f acc = {};
#pragma unroll
  for (int kk = 0; kk < 8; ++kk) {
    int d0 = kk * 32;
    v16bf ah, al, bh, bl;
    split_tile(arow + d0, arow + d0 + 8, ah, al);
    split_tile(brow + d0, brow + d0 + 8,  bh, bl);
    acc = wmma_bf16(al, bh, acc);
    acc = wmma_bf16(ah, bl, acc);
    acc = wmma_bf16(ah, bh, acc);
  }

  int n  = lane & 15;
  int mh = (lane >> 4) << 3;
#pragma unroll
  for (int r = 0; r < 8; ++r) {
    float x = acc[r] * scale;
    size_t idx = (size_t)(m0 + mh + r) * D_DIM + n0 + n;
    __bf16 hb = (__bf16)x;
    hi_plane[idx] = hb;
    lo_plane[idx] = (__bf16)(x - (float)hb);
  }
}

// ============================================================================
// Kernel 2: attention scores. Block = (b, 16-row L tile); wave = head.
// TDM double-buffers 4 k-planes (16 rows x 256 cols bf16) into padded LDS,
// heads read conflict-free ds_load_b128 B-fragments; 6 WMMA per 16x16 S-tile.
// ============================================================================
__global__ __launch_bounds__(256) void attn_kernel(
    const int* __restrict__ som, const float* __restrict__ mask,
    float* __restrict__ out, const __bf16* __restrict__ ws) {
  int lane = threadIdx.x & 31;
  int h    = threadIdx.x >> 5;       // 8 waves = 8 heads
  int b    = blockIdx.x >> 7;        // 512 blocks = 4 * 128
  int l0   = (blockIdx.x & 127) << 4;

  const __bf16* qsh = ws;
  const __bf16* qsl = ws + PLANE;
  const __bf16* qoh = ws + 2 * PLANE;
  const __bf16* qol = ws + 3 * PLANE;
  const __bf16* ksh = ws + 4 * PLANE;
  const __bf16* ksl = ws + 5 * PLANE;
  const __bf16* koh = ws + 6 * PLANE;
  const __bf16* kol = ws + 7 * PLANE;

  size_t qoff = (size_t)(b * L_DIM + l0) * D_DIM + h * HD;
  v16bf aSh = load_bf_a(qsh + qoff, lane);
  v16bf aSl = load_bf_a(qsl + qoff, lane);
  v16bf aOh = load_bf_a(qoh + qoff, lane);
  v16bf aOl = load_bf_a(qol + qoff, lane);

  int n  = lane & 15;
  int mh = (lane >> 4) << 3;

#if HAVE_TDM
  __shared__ __align__(16) unsigned char kstage[2][4][PLANE_LDS];
  const __bf16* kplane[4] = {ksh, ksl, koh, kol};
  const bool issuer = (threadIdx.x < 32);   // wave 0 drives the TDM

  if (issuer) {
    size_t roff = (size_t)(b * S_DIM) * D_DIM;
#pragma unroll
    for (int p = 0; p < 4; ++p)
      tdm_load_tile((unsigned)(uintptr_t)&kstage[0][p][0], kplane[p] + roff);
  }

  int cur = 0;
  for (int st = 0; st < S_TILES; ++st) {
    if (issuer) __builtin_amdgcn_s_wait_tensorcnt(0);  // buf[cur] landed
    __syncthreads();                                   // all see it; prev buf free
    if (issuer && (st + 1) < S_TILES) {
      size_t roff = (size_t)(b * S_DIM + (st + 1) * 16) * D_DIM;
#pragma unroll
      for (int p = 0; p < 4; ++p)
        tdm_load_tile((unsigned)(uintptr_t)&kstage[cur ^ 1][p][0], kplane[p] + roff);
    }

    int s0 = st << 4;
    v16bf bSh = lds_b_frag(&kstage[cur][0][0], lane, h);
    v16bf bSl = lds_b_frag(&kstage[cur][1][0], lane, h);
    v16bf bOh = lds_b_frag(&kstage[cur][2][0], lane, h);
    v16bf bOl = lds_b_frag(&kstage[cur][3][0], lane, h);
#else
  for (int st = 0; st < S_TILES; ++st) {
    int s0 = st << 4;
    size_t koff = (size_t)(b * S_DIM + s0) * D_DIM + h * HD;
    v16bf bSh = load_bf_b(ksh + koff, lane);
    v16bf bSl = load_bf_b(ksl + koff, lane);
    v16bf bOh = load_bf_b(koh + koff, lane);
    v16bf bOl = load_bf_b(kol + koff, lane);
#endif

    v8f accS = {};
    accS = wmma_bf16(aSl, bSh, accS);
    accS = wmma_bf16(aSh, bSl, accS);
    accS = wmma_bf16(aSh, bSh, accS);
    v8f accO = {};
    accO = wmma_bf16(aOl, bOh, accO);
    accO = wmma_bf16(aOh, bOl, accO);
    accO = wmma_bf16(aOh, bOh, accO);

#pragma unroll
    for (int r = 0; r < 8; ++r) {
      int l = l0 + mh + r;
      size_t mi = (size_t)(b * L_DIM + l) * S_DIM + s0 + n;   // (b,l,s)
      float res = (som[mi] != 0 ? accS[r] : accO[r]) + mask[mi];
      size_t oi = ((size_t)((b * H_NUM + h) * L_DIM + l)) * S_DIM + s0 + n;
      __builtin_nontemporal_store(res, out + oi);             // streamed output
    }
#if HAVE_TDM
    cur ^= 1;
#endif
  }
}

extern "C" void kernel_launch(void* const* d_in, const int* in_sizes, int n_in,
                              void* d_out, int out_size, void* d_ws, size_t ws_size,
                              hipStream_t stream) {
  const float* q    = (const float*)d_in[0];
  const float* kmat = (const float*)d_in[1];
  const int*   som  = (const int*)d_in[2];   // self_other_mask
  const float* mask = (const float*)d_in[3];
  const float* Wqs  = (const float*)d_in[4];
  const float* Wqo  = (const float*)d_in[5];
  const float* Wks  = (const float*)d_in[6];
  const float* Wko  = (const float*)d_in[7];
  float* out = (float*)d_out;
  __bf16* ws = (__bf16*)d_ws;   // needs 8 * PLANE * 2B = 32 MB scratch

  proj_kernel<<<dim3(4096), dim3(256), 0, stream>>>(q, kmat, Wqs, Wqo, Wks, Wko, ws);
  attn_kernel<<<dim3(512), dim3(256), 0, stream>>>(som, mask, out, ws);
}
